// ConvKAN3D_86998857548339
// MI455X (gfx1250) — matrix operand order repaired and checked by reading, not compiled, not run on hardware
//
#include <hip/hip_runtime.h>
#include <hip/hip_bf16.h>

typedef __attribute__((ext_vector_type(16))) _Float16 v16h;
typedef __attribute__((ext_vector_type(8)))  float    v8f;

// ---------------------------------------------------------------------------
// Weight packing: [COUT][CIN][27] f32 -> WMMA 16x16x32 f16 A-fragment layout,
// with the GEMM K axis reordered as k = tap*CIN + cin (tap = kd*9+kh*3+kw).
// Fragment layout (16-bit A 16x32, contiguous k order used here, same as B):
//   wp[((tm*KT + kstep)*32 + lane)*16 + e]
//   row m   = tm*16 + (lane&15)
//   k       = kstep*32 + e + ((lane&16) ? 16 : 0)
//   source  = w[m*K + (k%CIN)*27 + (k/CIN)]
// ---------------------------------------------------------------------------
template<int CIN, int COUT>
__global__ void pack_weights_kernel(const float* __restrict__ w,
                                    _Float16* __restrict__ wp) {
    constexpr int K  = CIN * 27;
    constexpr int KT = (K + 31) / 32;
    constexpr int TOTAL = (COUT / 16) * KT * 32 * 16;
    int idx = blockIdx.x * blockDim.x + threadIdx.x;
    if (idx >= TOTAL) return;
    int e     = idx & 15;
    int lane  = (idx >> 4) & 31;
    int kstep = (idx >> 9) % KT;
    int tm    = idx / (512 * KT);
    int m = tm * 16 + (lane & 15);
    int k = kstep * 32 + e + ((lane & 16) ? 16 : 0);
    float v = 0.0f;
    if (k < K) {
        int cin = k % CIN;
        int tap = k / CIN;
        v = w[m * K + cin * 27 + tap];
    }
    wp[idx] = (_Float16)v;
}

// ---------------------------------------------------------------------------
// Implicit-GEMM 3D conv (stride 1, pad 1, 3x3x3), batch N=2, cube size DIM.
// GEMM: M = COUT, N = flattened (n,d,h,w), K = 27*CIN (tap-major order).
// DIM is compile-time (power of two) so all spatial decode is shifts/masks
// and all strided loads/stores use immediate 24-bit instruction offsets.
// One wave owns 16 spatial columns x (COUT/MSPLIT) output channels:
// per K-step it builds ONE B fragment (uniform tap -> one bounds check,
// 16 loads at immediate offsets i*DHW*4) and issues TMW WMMAs against
// contiguous pre-packed A fragments.
// Grids are exact (M, S multiples of 16) so EXEC is all-1s at WMMA.
// ---------------------------------------------------------------------------
template<int CIN, int COUT, int MSPLIT, int DIM>
__global__ void __launch_bounds__(256)
conv3d_wmma_kernel(const float* __restrict__ x,        // [2][CIN][DIM^3]
                   const _Float16* __restrict__ wp,    // packed A fragments
                   const float* __restrict__ bias,     // [COUT]
                   float* __restrict__ y) {            // [2][COUT][DIM^3]
    constexpr int K   = CIN * 27;
    constexpr int KT  = (K + 31) / 32;
    constexpr int TM  = COUT / 16;
    constexpr int TMW = TM / MSPLIT;   // M-tiles per wave
    constexpr int D = DIM, H = DIM, W = DIM;
    constexpr int DHW = DIM * DIM * DIM;
    constexpr int S   = 2 * DHW;

    const int wave = (blockIdx.x * blockDim.x + threadIdx.x) >> 5;
    const int lane = threadIdx.x & 31;
    const int mg   = wave % MSPLIT;          // which M-group
    const int ts   = wave / MSPLIT;          // which spatial tile
    if (ts * 16 >= S) return;                // wave-uniform (grid is exact)
    const int tmbase = mg * TMW;

    const int hi  = (lane >> 4) & 1;
    const int col = lane & 15;

    // Decode this lane's spatial column (all shifts/masks: DIM is 2^n).
    const int s     = ts * 16 + col;
    const int n_idx = s / DHW;
    const int rem   = s & (DHW - 1);
    const int d0    = rem / (H * W);
    const int h0    = (rem / W) & (H - 1);
    const int w0    = rem & (W - 1);
    const float* xb = x + (long)n_idx * CIN * DHW;

    v8f acc[TMW] = {};

    for (int kstep = 0; kstep < KT; ++kstep) {
        // ---- Build B fragment (32 K-values x 16 spatial columns) ----
        v16h b;
        if constexpr (CIN % 32 == 0) {
            // One uniform tap (or half-tap) per K-step: k = tap*CIN + cin.
            const int tap     = (kstep * 32) / CIN;
            const int cinbase = (kstep * 32) % CIN + (hi ? 16 : 0);
            const int kd = tap / 9;  const int r  = tap - kd * 9;
            const int kh = r / 3;    const int kw = r - kh * 3;
            const int id = d0 + kd - 1, ih = h0 + kh - 1, iw = w0 + kw - 1;
            const bool valid = (unsigned)id < (unsigned)D &&
                               (unsigned)ih < (unsigned)H &&
                               (unsigned)iw < (unsigned)W;
            const float* p = xb + (long)cinbase * DHW + ((id * H + ih) * W + iw);
            #pragma unroll
            for (int i = 0; i < 16; ++i) {
                float v = valid ? p[i * DHW] : 0.0f;   // immediate offset
                b[i] = (_Float16)v;
            }
        } else {
            // CIN == 1: K = 27 (single padded K-step), per-element tap decode.
            #pragma unroll
            for (int i = 0; i < 16; ++i) {
                int k = kstep * 32 + i + (hi ? 16 : 0);
                float v = 0.0f;
                if (k < K) {
                    int kd = k / 9;  int r  = k - kd * 9;
                    int kh = r / 3;  int kw = r - kh * 3;
                    int id = d0 + kd - 1, ih = h0 + kh - 1, iw = w0 + kw - 1;
                    if ((unsigned)id < (unsigned)D &&
                        (unsigned)ih < (unsigned)H &&
                        (unsigned)iw < (unsigned)W)
                        v = xb[(id * H + ih) * W + iw];
                }
                b[i] = (_Float16)v;
            }
        }

        // Prefetch next K-step's A fragment region (global_prefetch_b8).
        if (kstep + 1 < KT)
            __builtin_prefetch(wp + ((long)(tmbase * KT + kstep + 1) * 32 + lane) * 16, 0, 3);

        // ---- TMW WMMAs sharing this B fragment ----
        #pragma unroll
        for (int t = 0; t < TMW; ++t) {
            v16h a = *(const v16h*)(wp +
                ((long)((tmbase + t) * KT + kstep) * 32 + lane) * 16);
            acc[t] = __builtin_amdgcn_wmma_f32_16x16x32_f16(
                /*neg_a=*/false, a, /*neg_b=*/false, b,
                /*c_mod=*/(short)0, acc[t], /*reuse_a=*/false, /*reuse_b=*/false);
        }
    }

    // D layout: element r -> row (r + hi*8), col = lane&15.
    // One base pointer per M-tile; 8 stores at immediate offsets r*DHW*4.
    #pragma unroll
    for (int t = 0; t < TMW; ++t) {
        const int mbase = (tmbase + t) * 16 + hi * 8;
        float* yp = y + (long)(n_idx * COUT + mbase) * DHW + rem;
        const float* bp = bias + mbase;
        #pragma unroll
        for (int r = 0; r < 8; ++r)
            yp[r * DHW] = acc[t][r] + bp[r];
    }
}

// ---------------------------------------------------------------------------
// Fused spline(KAN) + SiLU blend + eval-BN affine + 2x2x2 maxpool.
// One thread per pooled output element; DIM (input cube) is compile-time so
// the 8-element window loads use immediate offsets from one base pointer.
// ---------------------------------------------------------------------------
template<int C, int DIM>
__global__ void __launch_bounds__(256)
spline_bn_pool_kernel(const float* __restrict__ y,     // [2][C][DIM^3]
                      const float* __restrict__ knots, // [10]
                      const float* __restrict__ sw,    // [C][10]
                      const float* __restrict__ w1,
                      const float* __restrict__ w2,
                      const float* __restrict__ g,
                      const float* __restrict__ beta,
                      float* __restrict__ out) {       // [2][C][(DIM/2)^3]
    constexpr int H = DIM, W = DIM;
    constexpr int Dp = DIM / 2, Hp = DIM / 2, Wp = DIM / 2;
    constexpr int PHW = Dp * Hp * Wp;
    constexpr long TOTAL = 2L * C * PHW;
    long idx = (long)blockIdx.x * blockDim.x + threadIdx.x;
    if (idx >= TOTAL) return;

    const int wq = (int)(idx & (Wp - 1));
    const int hq = (int)((idx / Wp) & (Hp - 1));
    const int dq = (int)((idx / (Wp * Hp)) & (Dp - 1));
    const int c  = (int)((idx / PHW) % C);
    const int n  = (int)(idx / ((long)C * PHW));

    float kn[10], swc[10];
    #pragma unroll
    for (int i = 0; i < 10; ++i) { kn[i] = knots[i]; swc[i] = sw[c * 10 + i]; }
    const float scale = g[c] * rsqrtf(1.0f + 1e-5f);
    const float w1c = w1[c], w2c = w2[c], bc = beta[c];

    const float* yb = y + (long)(n * C + c) * (DIM * DIM * DIM)
                        + ((2 * dq * H + 2 * hq) * W + 2 * wq);
    float mx = -INFINITY;
    #pragma unroll
    for (int dz = 0; dz < 2; ++dz)
    #pragma unroll
    for (int dy = 0; dy < 2; ++dy)
    #pragma unroll
    for (int dx = 0; dx < 2; ++dx) {
        float v = yb[(dz * H + dy) * W + dx];          // immediate offset
        float sp = 0.0f;
        #pragma unroll
        for (int i = 0; i < 10; ++i) {
            float u = v - kn[i];
            u = u > 0.0f ? u : 0.0f;
            sp += u * u * u * swc[i];
        }
        float sil = v / (1.0f + __expf(-v));           // y * sigmoid(y)
        float o = (w1c * sp + w2c * sil) * scale + bc;
        mx = fmaxf(mx, o);
    }
    out[idx] = mx;
}

// ---------------------------------------------------------------------------
// Head: global mean over [8][8][8] -> fc1(128->256)+relu -> fc2(256->2).
// Single block of 256 threads; tiny.
// ---------------------------------------------------------------------------
__global__ void __launch_bounds__(256)
head_kernel(const float* __restrict__ pooled,  // [2][128][8][8][8]
            const float* __restrict__ fc1w,    // [256][128]
            const float* __restrict__ fc1b,    // [256]
            const float* __restrict__ fc2w,    // [2][256]
            const float* __restrict__ fc2b,    // [2]
            float* __restrict__ out) {         // [2][2]
    __shared__ float mean[256];
    __shared__ float hid[512];
    const int tid = threadIdx.x;

    // Stage 1: per-(n,c) mean over 512 spatial elements.
    {
        const float* p = pooled + (long)tid * 512;
        float s = 0.0f;
        for (int i = 0; i < 512; ++i) s += p[i];
        mean[tid] = s * (1.0f / 512.0f);
    }
    __syncthreads();

    // Stage 2: fc1 + relu. Thread tid owns hidden unit j=tid for both batches.
    for (int n = 0; n < 2; ++n) {
        float s = fc1b[tid];
        for (int c = 0; c < 128; ++c) s += mean[n * 128 + c] * fc1w[tid * 128 + c];
        hid[n * 256 + tid] = s > 0.0f ? s : 0.0f;
    }
    __syncthreads();

    // Stage 3: fc2.
    if (tid < 4) {
        int n = tid >> 1, o = tid & 1;
        float s = fc2b[o];
        for (int j = 0; j < 256; ++j) s += hid[n * 256 + j] * fc2w[o * 256 + j];
        out[n * 2 + o] = s;
    }
}

// ---------------------------------------------------------------------------
// Launch
// ---------------------------------------------------------------------------
extern "C" void kernel_launch(void* const* d_in, const int* in_sizes, int n_in,
                              void* d_out, int out_size, void* d_ws, size_t ws_size,
                              hipStream_t stream) {
    (void)in_sizes; (void)n_in; (void)out_size; (void)ws_size;

    const float* x     = (const float*)d_in[0];
    const float* c1_w  = (const float*)d_in[1];
    const float* c1_b  = (const float*)d_in[2];
    const float* c1_kn = (const float*)d_in[3];
    const float* c1_sw = (const float*)d_in[4];
    const float* c1_w1 = (const float*)d_in[5];
    const float* c1_w2 = (const float*)d_in[6];
    const float* bn1_g = (const float*)d_in[7];
    const float* bn1_b = (const float*)d_in[8];
    const float* c2_w  = (const float*)d_in[9];
    const float* c2_b  = (const float*)d_in[10];
    const float* c2_kn = (const float*)d_in[11];
    const float* c2_sw = (const float*)d_in[12];
    const float* c2_w1 = (const float*)d_in[13];
    const float* c2_w2 = (const float*)d_in[14];
    const float* bn2_g = (const float*)d_in[15];
    const float* bn2_b = (const float*)d_in[16];
    const float* c3_w  = (const float*)d_in[17];
    const float* c3_b  = (const float*)d_in[18];
    const float* c3_kn = (const float*)d_in[19];
    const float* c3_sw = (const float*)d_in[20];
    const float* c3_w1 = (const float*)d_in[21];
    const float* c3_w2 = (const float*)d_in[22];
    const float* bn3_g = (const float*)d_in[23];
    const float* bn3_b = (const float*)d_in[24];
    const float* fc1_w = (const float*)d_in[25];
    const float* fc1_b = (const float*)d_in[26];
    const float* fc2_w = (const float*)d_in[27];
    const float* fc2_b = (const float*)d_in[28];
    float* out = (float*)d_out;

    // Workspace layout (bytes):
    //   [0, 67108864)             y buffer (max conv output, stage 1)
    //   [67108864, 75497472)      pooled buffer (max pooled output, stage 1)
    //   [75497472, ...)           packed f16 weights (wp1, wp2, wp3)
    char* ws = (char*)d_ws;
    float*    ybuf = (float*)ws;
    float*    pbuf = (float*)(ws + 67108864);
    _Float16* wp1  = (_Float16*)(ws + 75497472);
    _Float16* wp2  = wp1 + 1024;      // 2*1*512 halves
    _Float16* wp3  = wp2 + 55296;     // 4*27*512 halves

    // --- Pack conv weights into WMMA A-fragment layout (tap-major K) ---
    pack_weights_kernel<1, 32> <<<(1024   + 255) / 256, 256, 0, stream>>>(c1_w, wp1);
    pack_weights_kernel<32, 64><<<(55296  + 255) / 256, 256, 0, stream>>>(c2_w, wp2);
    pack_weights_kernel<64,128><<<(221184 + 255) / 256, 256, 0, stream>>>(c3_w, wp3);

    // --- Block 1: conv 1->32 @ 64^3, spline+bn+pool -> 32^3 ---
    {
        constexpr int S = 2 * 64 * 64 * 64;             // 524288
        constexpr int waves = 1 * (S / 16);             // 32768 (MSPLIT=1)
        conv3d_wmma_kernel<1, 32, 1, 64><<<waves / 8, 256, 0, stream>>>(
            x, wp1, c1_b, ybuf);
        constexpr long total = 2L * 32 * 32 * 32 * 32;  // 2097152
        spline_bn_pool_kernel<32, 64><<<(unsigned)((total + 255) / 256), 256, 0, stream>>>(
            ybuf, c1_kn, c1_sw, c1_w1, c1_w2, bn1_g, bn1_b, pbuf);
    }

    // --- Block 2: conv 32->64 @ 32^3, spline+bn+pool -> 16^3 ---
    {
        constexpr int S = 2 * 32 * 32 * 32;             // 65536
        constexpr int waves = 1 * (S / 16);             // 4096 (MSPLIT=1)
        conv3d_wmma_kernel<32, 64, 1, 32><<<waves / 8, 256, 0, stream>>>(
            pbuf, wp2, c2_b, ybuf);
        constexpr long total = 2L * 64 * 16 * 16 * 16;  // 524288
        spline_bn_pool_kernel<64, 32><<<(unsigned)((total + 255) / 256), 256, 0, stream>>>(
            ybuf, c2_kn, c2_sw, c2_w1, c2_w2, bn2_g, bn2_b, pbuf);
    }

    // --- Block 3: conv 64->128 @ 16^3, spline+bn+pool -> 8^3 ---
    {
        constexpr int S = 2 * 16 * 16 * 16;             // 8192
        constexpr int waves = 2 * (S / 16);             // 1024 (MSPLIT=2)
        conv3d_wmma_kernel<64, 128, 2, 16><<<waves / 8, 256, 0, stream>>>(
            pbuf, wp3, c3_b, ybuf);
        constexpr long total = 2L * 128 * 8 * 8 * 8;    // 131072
        spline_bn_pool_kernel<128, 16><<<(unsigned)((total + 255) / 256), 256, 0, stream>>>(
            ybuf, c3_kn, c3_sw, c3_w1, c3_w2, bn3_g, bn3_b, pbuf);
    }

    // --- Head: mean pool + fc1 + relu + fc2 ---
    head_kernel<<<1, 256, 0, stream>>>(pbuf, fc1_w, fc1_b, fc2_w, fc2_b, out);
}
